// QuantizableBidirectionalPropagation_51685636440891
// MI455X (gfx1250) — matrix operand order repaired and checked by reading, not compiled, not run on hardware
//
#include <hip/hip_runtime.h>
#include <hip/hip_bf16.h>

// Problem constants (B=1 throughout)
#define HH 96
#define WW 96
#define PLANE (HH * WW)          // 9216, divisible by 16
#define CHN 64
#define F64 ((size_t)CHN * PLANE)
#define TT 8
#define DGRP 16
#define MAX_RES_F 10.0f

typedef __attribute__((ext_vector_type(16))) _Float16 v16h;
typedef __attribute__((ext_vector_type(8)))  float    v8f;

__device__ __forceinline__ int clampi(int v, int lo, int hi) {
    return v < lo ? lo : (v > hi ? hi : v);
}

// ---------------------------------------------------------------------------
// Weight pack: f32 [Cout][K] -> f16 [S][Cout][32], S = ceil(K/32), zero-padded.
// Within each 32-block, position pos = half*16 + e stores
//   K = s*32 + half*16 + e
// so a lane's 16 A-fragment elements are contiguous halfs.
// ---------------------------------------------------------------------------
__global__ void pack_weights(const float* __restrict__ W, _Float16* __restrict__ out,
                             int K, int Cout)
{
    int S = (K + 31) >> 5;
    int total = S * Cout * 32;
    int idx = blockIdx.x * blockDim.x + threadIdx.x;
    if (idx >= total) return;
    int pos = idx & 31;
    int m   = (idx >> 5) % Cout;
    int s   = idx / (Cout * 32);
    int half = pos >> 4, e = pos & 15;
    int k = (s << 5) + (half << 4) + e;
    out[idx] = (_Float16)((k < K) ? W[(size_t)m * K + k] : 0.f);
}

// ---------------------------------------------------------------------------
// Im2col pack: X [Cin][HH][WW] f32 -> B f16 [S][PLANE][32] in plain K order,
// matching the contiguous order of the packed A operand (lane: K = half*16 + e).
// ---------------------------------------------------------------------------
__global__ void im2col_pack(const float* __restrict__ X, _Float16* __restrict__ out,
                            int Cin, int taps)
{
    int K = Cin * taps;
    int S = (K + 31) >> 5;
    size_t total = (size_t)S * PLANE * 32;
    size_t idx = blockIdx.x * (size_t)blockDim.x + threadIdx.x;
    if (idx >= total) return;
    int kl = (int)(idx & 31);
    int p  = (int)((idx >> 5) % PLANE);
    int s  = (int)(idx / ((size_t)PLANE * 32));
    int k  = (s << 5) + kl;
    float v = 0.f;
    if (k < K) {
        int ci = k / taps;
        int r  = k - ci * taps;
        int yy = p / WW, xx = p % WW;
        if (taps == 9) { yy += (r / 3) - 1; xx += (r % 3) - 1; }
        if ((unsigned)yy < (unsigned)HH && (unsigned)xx < (unsigned)WW)
            v = X[(size_t)ci * PLANE + yy * WW + xx];
    }
    out[idx] = (_Float16)v;
}

// ---------------------------------------------------------------------------
// Multi-M-tile, double-buffered WMMA GEMM on packed operands.
// One wave -> MT vertically-stacked 16x16 tiles sharing one B fragment.
// Ping-pong fragment registers: loads for step s+1 are issued before the
// WMMAs of step s, so v_wmma overlaps the next global_load_b128 group.
// Epilogue: +bias, optional lrelu, optional elementwise add of addSrc.
// ---------------------------------------------------------------------------
template <int MT>
__global__ __launch_bounds__(64) void gemm_packed_wmma(
    const _Float16* __restrict__ Ap,  // [S][Cout][32] packed weights
    const _Float16* __restrict__ Bp,  // [S][PLANE][32] packed im2col / dcn vals
    const float* __restrict__ Bs,     // [Cout]
    float* __restrict__ Y,            // [Cout][PLANE]
    int S, int Cout, int act,
    const float* __restrict__ addSrc) // nullable: Y += addSrc (elementwise)
{
    const int lane = threadIdx.x & 31;
    const int tile = blockIdx.x * 2 + (threadIdx.x >> 5);
    const int nTiles = PLANE >> 4;                 // 576
    const int mGroups = (Cout >> 4) / MT;
    if (tile >= mGroups * nTiles) return;          // wave-uniform exit

    const int mg = tile / nTiles;
    const int n0 = (tile % nTiles) << 4;
    const int half = lane >> 4;
    const int mn   = lane & 15;
    const int col  = n0 + mn;

    const size_t aStep = (size_t)Cout * 32;
    const size_t bStep = (size_t)PLANE * 32;
    const _Float16* aPtr = Ap + ((size_t)(mg * MT * 16 + mn)) * 32 + half * 16;
    const _Float16* bPtr = Bp + ((size_t)col) * 32 + half * 16;

    const v8f vzero = {0.f, 0.f, 0.f, 0.f, 0.f, 0.f, 0.f, 0.f};
    v8f acc[MT];
#pragma unroll
    for (int m = 0; m < MT; ++m) acc[m] = vzero;

    v16h a0[MT], a1[MT], b0, b1;
    // preload step 0
#pragma unroll
    for (int m = 0; m < MT; ++m)
        a0[m] = *(const v16h*)(aPtr + (size_t)m * 16 * 32);
    b0 = *(const v16h*)bPtr;
    aPtr += aStep;
    bPtr += bStep;

    int s = 0;
    for (; s + 2 <= S; s += 2) {
        if (s + 4 < S) __builtin_prefetch(bPtr + 3 * bStep, 0, 3);  // global_prefetch_b8
        // load step s+1 into ping buffers
#pragma unroll
        for (int m = 0; m < MT; ++m)
            a1[m] = *(const v16h*)(aPtr + (size_t)m * 16 * 32);
        b1 = *(const v16h*)bPtr;
        aPtr += aStep;
        bPtr += bStep;
        // MMA step s (overlaps the loads above)
#pragma unroll
        for (int m = 0; m < MT; ++m)
            acc[m] = __builtin_amdgcn_wmma_f32_16x16x32_f16(
                false, a0[m], false, b0, (short)0, acc[m], false, false);
        // load step s+2 into pong buffers
        if (s + 2 < S) {
#pragma unroll
            for (int m = 0; m < MT; ++m)
                a0[m] = *(const v16h*)(aPtr + (size_t)m * 16 * 32);
            b0 = *(const v16h*)bPtr;
            aPtr += aStep;
            bPtr += bStep;
        }
        // MMA step s+1
#pragma unroll
        for (int m = 0; m < MT; ++m)
            acc[m] = __builtin_amdgcn_wmma_f32_16x16x32_f16(
                false, a1[m], false, b1, (short)0, acc[m], false, false);
    }
    if (s < S) {   // odd-S tail (a0/b0 hold step S-1)
#pragma unroll
        for (int m = 0; m < MT; ++m)
            acc[m] = __builtin_amdgcn_wmma_f32_16x16x32_f16(
                false, a0[m], false, b0, (short)0, acc[m], false, false);
    }

#pragma unroll
    for (int m = 0; m < MT; ++m) {
#pragma unroll
        for (int r = 0; r < 8; ++r) {
            int co = mg * MT * 16 + m * 16 + (half << 3) + r;
            float v = acc[m][r] + Bs[co];
            if (act) v = (v >= 0.f) ? v : 0.1f * v;    // lrelu(0.1)
            if (addSrc) v += addSrc[(size_t)co * PLANE + col];
            Y[(size_t)co * PLANE + col] = v;
        }
    }
}

// ---------------------------------------------------------------------------
// Bilinear flow warp: out[c][p] = sample(feat[c], x + flow_x, y + flow_y)
// ---------------------------------------------------------------------------
__global__ void warp_kernel(const float* __restrict__ feat,
                            const float* __restrict__ flow,
                            float* __restrict__ out, int C)
{
    int p = blockIdx.x * blockDim.x + threadIdx.x;
    if (p >= PLANE) return;
    float fx = (float)(p % WW) + flow[p];
    float fy = (float)(p / WW) + flow[PLANE + p];
    float x0f = floorf(fx), y0f = floorf(fy);
    float wx = fx - x0f, wy = fy - y0f;
    int x0 = (int)x0f, y0 = (int)y0f, x1 = x0 + 1, y1 = y0 + 1;
    float vx0 = (x0 >= 0 && x0 < WW) ? 1.f : 0.f;
    float vx1 = (x1 >= 0 && x1 < WW) ? 1.f : 0.f;
    float vy0 = (y0 >= 0 && y0 < HH) ? 1.f : 0.f;
    float vy1 = (y1 >= 0 && y1 < HH) ? 1.f : 0.f;
    int cx0 = clampi(x0, 0, WW - 1), cx1 = clampi(x1, 0, WW - 1);
    int cy0 = clampi(y0, 0, HH - 1), cy1 = clampi(y1, 0, HH - 1);
    int i00 = cy0 * WW + cx0, i10 = cy0 * WW + cx1;
    int i01 = cy1 * WW + cx0, i11 = cy1 * WW + cx1;
    float w00 = (1.f - wx) * (1.f - wy) * vx0 * vy0;
    float w10 = wx * (1.f - wy) * vx1 * vy0;
    float w01 = (1.f - wx) * wy * vx0 * vy1;
    float w11 = wx * wy * vx1 * vy1;
    for (int c = 0; c < C; ++c) {
        const float* ch = feat + (size_t)c * PLANE;
        out[(size_t)c * PLANE + p] =
            w00 * ch[i00] + w10 * ch[i10] + w01 * ch[i01] + w11 * ch[i11];
    }
}

// ---------------------------------------------------------------------------
// Offset/mask prep: h[0:288] -> offset = 10*tanh + tiled reversed flow,
// h[288:432] -> mask = sigmoid.
// ---------------------------------------------------------------------------
__global__ void offset_prep(const float* __restrict__ h,
                            const float* __restrict__ f1,
                            const float* __restrict__ f2,
                            float* __restrict__ off,
                            float* __restrict__ msk)
{
    size_t idx = blockIdx.x * (size_t)blockDim.x + threadIdx.x;
    if (idx >= (size_t)432 * PLANE) return;
    int c = (int)(idx / PLANE);
    int p = (int)(idx % PLANE);
    float v = h[idx];
    if (c < 288) {
        float o = MAX_RES_F * tanhf(v);
        const float* fl = (c < 144) ? f1 : f2;
        int cc = c % 144;
        // tile(flow[:, ::-1]): even channel -> flow ch1 (y), odd -> flow ch0 (x)
        o += fl[(size_t)((cc & 1) ^ 1) * PLANE + p];
        off[idx] = o;
    } else {
        msk[(size_t)(c - 288) * PLANE + p] = 1.f / (1.f + expf(-v));
    }
}

// ---------------------------------------------------------------------------
// Deformable sampling, writing directly into the packed B layout:
// Bp[k>>5][p][k&31] = mask * bilinear(Xcat[ci]) with k = ci*9 + tap, ci = g*8+c.
// ---------------------------------------------------------------------------
__global__ void dcn_sample(const float* __restrict__ Xc,    // [128][PLANE]
                           const float* __restrict__ off,   // [288][PLANE]
                           const float* __restrict__ msk,   // [144][PLANE]
                           _Float16* __restrict__ Bp)       // [36][PLANE][32]
{
    size_t idx = blockIdx.x * (size_t)blockDim.x + threadIdx.x;
    if (idx >= (size_t)DGRP * 9 * PLANE) return;
    int p  = (int)(idx % PLANE);
    int gt = (int)(idx / PLANE);   // g*9 + tap
    int tap = gt % 9;
    int g   = gt / 9;
    float oy = off[(size_t)(gt * 2 + 0) * PLANE + p];
    float ox = off[(size_t)(gt * 2 + 1) * PLANE + p];
    float m  = msk[(size_t)gt * PLANE + p];
    float fy = oy + (float)(tap / 3 - 1) + (float)(p / WW);
    float fx = ox + (float)(tap % 3 - 1) + (float)(p % WW);
    float x0f = floorf(fx), y0f = floorf(fy);
    float wx = fx - x0f, wy = fy - y0f;
    int x0 = (int)x0f, y0 = (int)y0f, x1 = x0 + 1, y1 = y0 + 1;
    float vx0 = (x0 >= 0 && x0 < WW) ? 1.f : 0.f;
    float vx1 = (x1 >= 0 && x1 < WW) ? 1.f : 0.f;
    float vy0 = (y0 >= 0 && y0 < HH) ? 1.f : 0.f;
    float vy1 = (y1 >= 0 && y1 < HH) ? 1.f : 0.f;
    int cx0 = clampi(x0, 0, WW - 1), cx1 = clampi(x1, 0, WW - 1);
    int cy0 = clampi(y0, 0, HH - 1), cy1 = clampi(y1, 0, HH - 1);
    int i00 = cy0 * WW + cx0, i10 = cy0 * WW + cx1;
    int i01 = cy1 * WW + cx0, i11 = cy1 * WW + cx1;
    float w00 = (1.f - wx) * (1.f - wy) * vx0 * vy0;
    float w10 = wx * (1.f - wy) * vx1 * vy0;
    float w01 = (1.f - wx) * wy * vx0 * vy1;
    float w11 = wx * wy * vx1 * vy1;
#pragma unroll
    for (int c = 0; c < 8; ++c) {
        int k = (g * 8 + c) * 9 + tap;
        const float* ch = Xc + (size_t)(g * 8 + c) * PLANE;
        float v = w00 * ch[i00] + w10 * ch[i10] + w01 * ch[i01] + w11 * ch[i11];
        Bp[((size_t)(k >> 5) * PLANE + p) * 32 + (k & 31)] = (_Float16)(m * v);
    }
}

__global__ void fill_kernel(float* __restrict__ p, size_t n)
{
    size_t i = blockIdx.x * (size_t)blockDim.x + threadIdx.x;
    if (i < n) p[i] = 0.f;
}

__global__ void add_kernel(const float* __restrict__ a, const float* __restrict__ b,
                           float* __restrict__ o, size_t n)
{
    size_t i = blockIdx.x * (size_t)blockDim.x + threadIdx.x;
    if (i < n) o[i] = a[i] + b[i];
}

// ---------------------------------------------------------------------------
// Host orchestration
// ---------------------------------------------------------------------------
struct DirRaw {
    const float *ocw[4], *ocb[4];
    const float *dcnw, *dcnb;
    const float *bw0, *bb0, *bw1, *bb1;
};
struct PackedW { const _Float16* p; int S; };

extern "C" void kernel_launch(void* const* d_in, const int* in_sizes, int n_in,
                              void* d_out, int out_size, void* d_ws, size_t ws_size,
                              hipStream_t stream)
{
    (void)in_sizes; (void)n_in; (void)out_size; (void)ws_size;

    const float* x   = (const float*)d_in[0];   // (1,8,64,96,96)
    const float* fbk = (const float*)d_in[1];   // (1,7,2,96,96)
    const float* ffw = (const float*)d_in[2];   // (1,7,2,96,96)
    int ip = 3;
    DirRaw R[2];
    for (int d = 0; d < 2; ++d) {
        for (int j = 0; j < 4; ++j) {
            R[d].ocw[j] = (const float*)d_in[ip++];
            R[d].ocb[j] = (const float*)d_in[ip++];
        }
        R[d].dcnw = (const float*)d_in[ip++];
        R[d].dcnb = (const float*)d_in[ip++];
        R[d].bw0  = (const float*)d_in[ip++];
        R[d].bb0  = (const float*)d_in[ip++];
        R[d].bw1  = (const float*)d_in[ip++];
        R[d].bb1  = (const float*)d_in[ip++];
    }
    const float* fusw = (const float*)d_in[ip++]; // (64,128,1,1)
    const float* fusb = (const float*)d_in[ip++];

    // 64B-aligned workspace bump allocator
    char* wsb = (char*)d_ws;
    size_t off = 0;
    auto allocb = [&](size_t bytes) {
        off = (off + 63) & ~(size_t)63;
        void* p = (void*)(wsb + off);
        off += bytes;
        return p;
    };
    auto allocf = [&](size_t nfloats) { return (float*)allocb(nfloats * sizeof(float)); };
    auto alloch = [&](size_t nhalf)   { return (_Float16*)allocb(nhalf * sizeof(_Float16)); };

    float* feats_b   = allocf((size_t)TT * F64);
    float* feats_f   = allocf((size_t)TT * F64);
    float* feat_prop = allocf(F64);
    float* cond_n1   = allocf(F64);
    float* cond_n2   = allocf(F64);
    float* zfeat     = allocf(F64);
    float* flow2b    = allocf(2 * PLANE);
    float* tmp2      = allocf(2 * PLANE);
    float* cond      = allocf((size_t)196 * PLANE);
    float* t64a      = allocf(F64);
    float* t64b      = allocf(F64);
    float* h432      = allocf((size_t)432 * PLANE);
    float* off288    = allocf((size_t)288 * PLANE);
    float* msk144    = allocf((size_t)144 * PLANE);
    float* fcat      = allocf((size_t)192 * PLANE);
    // shared packed-B buffer, sized for the largest K (196*9=1764 -> S=56)
    _Float16* bpack  = alloch((size_t)56 * PLANE * 32);

    // ---- pack all weights once (constant per launch, deterministic) ----
    auto PACKW = [&](const float* W, int K, int Cout) -> PackedW {
        int S = (K + 31) >> 5;
        _Float16* buf = alloch((size_t)S * Cout * 32);
        int total = S * Cout * 32;
        pack_weights<<<(total + 255) / 256, 256, 0, stream>>>(W, buf, K, Cout);
        PackedW r; r.p = buf; r.S = S; return r;
    };
    PackedW pw_oc[2][4], pw_dcn[2], pw_bb0[2], pw_bb1[2];
    for (int d = 0; d < 2; ++d) {
        pw_oc[d][0] = PACKW(R[d].ocw[0], 196 * 9, 64);
        pw_oc[d][1] = PACKW(R[d].ocw[1],  64 * 9, 64);
        pw_oc[d][2] = PACKW(R[d].ocw[2],  64 * 9, 64);
        pw_oc[d][3] = PACKW(R[d].ocw[3],  64 * 9, 432);
        pw_dcn[d]   = PACKW(R[d].dcnw,   128 * 9, 64);
        pw_bb0[d]   = PACKW(R[d].bw0, (d == 0 ? 128 : 192) * 9, 64);
        pw_bb1[d]   = PACKW(R[d].bw1,     64 * 9, 64);
    }
    PackedW pw_fus = PACKW(fusw, 128, 64);

    auto GEMM = [&](const PackedW& W, const _Float16* B, const float* Bs, float* Y,
                    int Cout, int act, const float* addSrc) {
        const int nT = PLANE >> 4;  // 576
        if (Cout == 432) {          // 27 m-blocks -> 9 groups of 3
            int tiles = ((432 >> 4) / 3) * nT;
            gemm_packed_wmma<3><<<(tiles + 1) / 2, 64, 0, stream>>>(
                W.p, B, Bs, Y, W.S, Cout, act, addSrc);
        } else {                    // Cout=64: 4 m-blocks -> 2 groups of 2
            int tiles = ((Cout >> 4) / 2) * nT;
            gemm_packed_wmma<2><<<(tiles + 1) / 2, 64, 0, stream>>>(
                W.p, B, Bs, Y, W.S, Cout, act, addSrc);
        }
    };
    auto CONVP = [&](const float* X, const PackedW& W, const float* Bs, float* Y,
                     int Cin, int Cout, int taps, int act, const float* addSrc) {
        size_t total = (size_t)W.S * PLANE * 32;
        im2col_pack<<<(unsigned)((total + 255) / 256), 256, 0, stream>>>(X, bpack, Cin, taps);
        GEMM(W, bpack, Bs, Y, Cout, act, addSrc);
    };
    auto FILL0 = [&](float* p, size_t n) {
        fill_kernel<<<(unsigned)((n + 255) / 256), 256, 0, stream>>>(p, n);
    };
    auto ADD = [&](const float* a, const float* b, float* o, size_t n) {
        add_kernel<<<(unsigned)((n + 255) / 256), 256, 0, stream>>>(a, b, o, n);
    };
    auto CPY = [&](float* dst, const float* src, size_t nfloats) {
        hipMemcpyAsync(dst, src, nfloats * sizeof(float), hipMemcpyDeviceToDevice, stream);
    };

    for (int dir = 0; dir < 2; ++dir) {
        const DirRaw& rp = R[dir];
        const float* flows = (dir == 0) ? fbk : ffw;
        float* feats = (dir == 0) ? feats_b : feats_f;

        FILL0(feat_prop, F64);
        FILL0(cond_n2, F64);
        FILL0(flow2b, 2 * PLANE);
        FILL0(zfeat, F64);

        for (int i = 0; i < TT; ++i) {
            int idx = (dir == 0) ? (TT - 1 - i) : i;
            const float* feat_cur = x + (size_t)idx * F64;

            if (i > 0) {
                const float* flow_n1 = flows + (size_t)(i - 1) * 2 * PLANE;
                warp_kernel<<<(PLANE + 255) / 256, 256, 0, stream>>>(feat_prop, flow_n1, cond_n1, CHN);
                const float* feat_n2 = zfeat;
                if (i > 1) {
                    int idx2 = (dir == 0) ? (idx + 2) : (idx - 2);
                    feat_n2 = feats + (size_t)idx2 * F64;
                    const float* flow_n2s = flows + (size_t)(i - 2) * 2 * PLANE;
                    warp_kernel<<<(PLANE + 255) / 256, 256, 0, stream>>>(flow_n2s, flow_n1, tmp2, 2);
                    ADD(flow_n1, tmp2, flow2b, 2 * PLANE);
                    warp_kernel<<<(PLANE + 255) / 256, 256, 0, stream>>>(feat_n2, flow2b, cond_n2, CHN);
                }
                // cond = concat(cond_n1, feat_cur, cond_n2, flow_n1, flow_n2) -> 196 ch
                CPY(cond, cond_n1, F64);
                CPY(cond + F64, feat_cur, F64);
                CPY(cond + 2 * F64, cond_n2, F64);
                CPY(cond + 3 * F64, flow_n1, 2 * PLANE);
                CPY(cond + 3 * F64 + 2 * PLANE, flow2b, 2 * PLANE);

                // offset conv stack
                CONVP(cond, pw_oc[dir][0], rp.ocb[0], t64a, 196, 64, 9, 1, nullptr);
                CONVP(t64a, pw_oc[dir][1], rp.ocb[1], t64b, 64, 64, 9, 1, nullptr);
                CONVP(t64b, pw_oc[dir][2], rp.ocb[2], t64a, 64, 64, 9, 1, nullptr);
                CONVP(t64a, pw_oc[dir][3], rp.ocb[3], h432, 64, 432, 9, 0, nullptr);
                offset_prep<<<(unsigned)(((size_t)432 * PLANE + 255) / 256), 256, 0, stream>>>(
                    h432, flow_n1, flow2b, off288, msk144);

                // modulated deformable conv: packed sampling then WMMA GEMM
                CPY(fcat, feat_prop, F64);
                CPY(fcat + F64, feat_n2, F64);
                dcn_sample<<<(unsigned)(((size_t)DGRP * 9 * PLANE + 255) / 256), 256, 0, stream>>>(
                    fcat, off288, msk144, bpack);
                GEMM(pw_dcn[dir], bpack, rp.dcnb, feat_prop, 64, 0, nullptr);
            }

            // backbone (second conv fuses the residual add into the epilogue)
            if (dir == 0) {
                CPY(fcat, feat_cur, F64);
                CPY(fcat + F64, feat_prop, F64);
                CONVP(fcat, pw_bb0[dir], rp.bb0, t64a, 128, 64, 9, 1, nullptr);
            } else {
                CPY(fcat, feat_cur, F64);
                CPY(fcat + F64, feats_b + (size_t)idx * F64, F64);
                CPY(fcat + 2 * F64, feat_prop, F64);
                CONVP(fcat, pw_bb0[dir], rp.bb0, t64a, 192, 64, 9, 1, nullptr);
            }
            CONVP(t64a, pw_bb1[dir], rp.bb1, feat_prop, 64, 64, 9, 0, feat_prop);
            CPY(feats + (size_t)idx * F64, feat_prop, F64);

            // fuse inline during forward pass: out[idx] = fusion(cat(b, f)) + x[idx]
            if (dir == 1) {
                float* outp = (float*)d_out + (size_t)idx * F64;
                CPY(fcat, feats_b + (size_t)idx * F64, F64);
                CPY(fcat + F64, feat_prop, F64);
                CONVP(fcat, pw_fus, fusb, outp, 128, 64, 1, 0, feat_cur);
            }
        }
    }
}
